// LSTMAttentionDecoder_80985903333634
// MI455X (gfx1250) — matrix operand order repaired and checked by reading, not compiled, not run on hardware
//
#include <hip/hip_runtime.h>
#include <hip/hip_bf16.h>
#include <math.h>

// ---------------------------------------------------------------------------
// LSTM + cross-attention greedy decoder for MI455X (gfx1250, wave32, WMMA).
//
// Design (roofline reasoning, compile-only):
//  * All weights converted once per launch to bf16 in the exact
//    V_WMMA_F32_16X16X32_BF16 B-matrix lane layout (~67 MB -> L2-resident,
//    192 MB L2). Each of the 128 sequential steps streams them from L2.
//  * K/V encoder projections computed once with an LDS-staged WMMA GEMM,
//    stored bf16 [B,H,S,DH].
//  * Skinny M=32 GEMMs: each wave owns one N-tile, computes both M-tiles
//    (B operand reused across the two WMMAs). Deterministic split-K
//    (gridDim.y partials, fixed-order combine in the consumer kernel) raises
//    block counts so more WGPs pull from L2 concurrently.
//  * Producers write the WMMA-A bf16 packed layout directly (gates/attention/
//    gemm packed output), so the 128-step critical path is 11 kernels/step.
// ---------------------------------------------------------------------------

typedef __bf16 bf16;
typedef __attribute__((ext_vector_type(16))) __bf16 v16bf;
typedef __attribute__((ext_vector_type(8)))  float  v8f;

#define BB  32
#define SS  1024
#define DD  1024
#define HH  16
#define VV  10000
#define TT  128
#define DHH 64

__device__ __forceinline__ float sigf(float x) { return 1.f / (1.f + __expf(-x)); }

__device__ __forceinline__ v8f wmma_bf16(v16bf a, v16bf b, v8f c) {
    return __builtin_amdgcn_wmma_f32_16x16x32_bf16(false, a, false, b, (short)0, c,
                                                   false, false);
}

// A-matrix (16x32 bf16) per-lane k for element e (0..15), half = lane>>4:
//   k = (e&7) + half*8 + (e>=8 ? 16 : 0)
__device__ __forceinline__ int a_klocal(int e, int half) {
    return (e & 7) + half * 8 + ((e >> 3) << 4);
}

// Scatter one f32 value (row m of 32, column k) into WMMA-A bf16 packed
// layout: dst[(kt*2+mt)*512 + lane*16 + e].  (Inverse of a_klocal.)
__device__ __forceinline__ void store_packed(bf16* dst, int m, int k, float v) {
    int kt   = k >> 5;
    int kl   = k & 31;
    int half = (kl >> 3) & 1;
    int e    = (kl & 7) | ((kl >> 4) << 3);
    int lane = half * 16 + (m & 15);
    int mt   = m >> 4;
    dst[((kt * 2 + mt) * 512) + lane * 16 + e] = (bf16)v;
}

// ---------------------------------------------------------------------------
// Pack A rows (f32, optional row-gather for embedding lookup) into WMMA-A
// bf16 layout. total = cols*32 elements; writes a contiguous K-major region.
// ---------------------------------------------------------------------------
__global__ void pack_a(const float* __restrict__ s0, const int* __restrict__ idx0,
                       int cols, bf16* __restrict__ out, int total) {
    int i = blockIdx.x * blockDim.x + threadIdx.x;
    if (i >= total) return;
    int e    = i & 15;
    int lane = (i >> 4) & 31;
    int mt   = (i >> 9) & 1;
    int kt   = i >> 10;
    int m    = mt * 16 + (lane & 15);
    int k    = kt * 32 + a_klocal(e, lane >> 4);
    const float* row = idx0 ? (s0 + (size_t)idx0[m] * cols) : (s0 + (size_t)m * cols);
    out[i] = (bf16)row[k];
}

// ---------------------------------------------------------------------------
// Pack W (row-major [N, K0(+K1 stacked along k)]) into WMMA-B bf16 layout:
// out[(nt*Ktiles+kt)*512 + lane*16 + e], k = kt*32 + (lane>>4)*8 + (e&7) + (e>>3)*16,
// n = nt*16 + (lane&15).
// ---------------------------------------------------------------------------
__global__ void pack_b(const float* __restrict__ W0, int K0,
                       const float* __restrict__ W1, int K1,
                       bf16* __restrict__ out, int Ktiles, long total) {
    long i = (long)blockIdx.x * blockDim.x + threadIdx.x;
    if (i >= total) return;
    int  e    = (int)(i & 15);
    int  lane = (int)((i >> 4) & 31);
    long tile = i >> 9;
    int  kt   = (int)(tile % Ktiles);
    int  nt   = (int)(tile / Ktiles);
    int  n    = nt * 16 + (lane & 15);
    int  k    = kt * 32 + (lane >> 4) * 8 + (e & 7) + ((e >> 3) << 4);
    float v = (k < K0) ? W0[(size_t)n * K0 + k] : W1[(size_t)n * K1 + (k - K0)];
    out[i] = (bf16)v;
}

// ---------------------------------------------------------------------------
// Skinny GEMM C[32,N] = A[32,K] * W^T.
//  ksplit>1 : blockIdx.y = K-chunk; raw f32 partials -> Cf[ks][32][N]
//             (consumer combines in fixed order; fully deterministic).
//  ksplit==1: +bias, optional tanh; writes either f32 C or directly the
//             WMMA-A packed bf16 of the *next* GEMM (Pout).
// ---------------------------------------------------------------------------
__global__ void gemm32(const bf16* __restrict__ Apk, const bf16* __restrict__ Wpk,
                       const float* __restrict__ bias, float* __restrict__ Cf,
                       bf16* __restrict__ Pout,
                       int N, int Ntiles, int Ktiles, int act, int ksplit) {
    int wid  = threadIdx.x >> 5;
    int lane = threadIdx.x & 31;
    int ks    = blockIdx.y;
    int ktper = Ktiles / ksplit;
    int kt0   = ks * ktper;
    const v16bf* Av = (const v16bf*)Apk;
    const v16bf* Wv = (const v16bf*)Wpk;
    for (int nt = blockIdx.x * 8 + wid; nt < Ntiles; nt += gridDim.x * 8) {
        v8f c0 = {}; v8f c1 = {};
        const v16bf* wp = Wv + (size_t)nt * Ktiles * 32 + lane;
        const v16bf* ap = Av + lane;
        for (int kt = kt0; kt < kt0 + ktper; ++kt) {
            v16bf b  = wp[(size_t)kt * 32];
            v16bf a0 = ap[(size_t)(kt * 2 + 0) * 32];
            v16bf a1 = ap[(size_t)(kt * 2 + 1) * 32];
            c0 = wmma_bf16(a0, b, c0);
            c1 = wmma_bf16(a1, b, c1);
        }
        int n  = nt * 16 + (lane & 15);
        int mh = (lane >> 4) * 8;
        if (ksplit > 1) {
            float* Cp = Cf + (size_t)ks * 32 * N;
            for (int r = 0; r < 8; ++r) {
                Cp[(size_t)(mh + r) * N + n]      = c0[r];
                Cp[(size_t)(mh + r + 16) * N + n] = c1[r];
            }
        } else {
            float bv = bias ? bias[n] : 0.f;
            for (int r = 0; r < 8; ++r) {
                float v0 = c0[r] + bv;
                float v1 = c1[r] + bv;
                if (act == 1) { v0 = tanhf(v0); v1 = tanhf(v1); }
                if (Pout) {
                    store_packed(Pout, mh + r, n, v0);
                    store_packed(Pout, mh + r + 16, n, v1);
                } else {
                    Cf[(size_t)(mh + r) * N + n]      = v0;
                    Cf[(size_t)(mh + r + 16) * N + n] = v1;
                }
            }
        }
    }
}

// ---------------------------------------------------------------------------
// One-time K/V projection: K = enc@wk^T+bk ; V = enc@wv^T+bv ; bf16 [B,H,S,DH].
// grid = (B*S/32, 16); LDS-staged A pack (8-ktile chunks, 16KB); wave-owned
// N-tile (blockIdx.y*8+wid: 0..63 -> K, 64..127 -> V).
// ---------------------------------------------------------------------------
__global__ void kv_gemm(const float* __restrict__ enc,
                        const bf16* __restrict__ Wk, const bf16* __restrict__ Wv,
                        const float* __restrict__ bk, const float* __restrict__ bv,
                        bf16* __restrict__ Ko, bf16* __restrict__ Vo) {
    __shared__ __align__(32) bf16 apk[8192];  // 8 ktiles * 2 mt * 512
    int tid = threadIdx.x;
    int wid = tid >> 5, lane = tid & 31;
    int mb = blockIdx.x;
    int nt = blockIdx.y * 8 + wid;            // 0..127
    const bf16* Wp = (nt < 64) ? Wk : Wv;
    int ntl = (nt < 64) ? nt : nt - 64;
    const v16bf* Wvp = (const v16bf*)Wp;
    v8f c0 = {}, c1 = {};
    for (int ck = 0; ck < 4; ++ck) {
        for (int i = tid; i < 8192; i += 256) {
            int e   = i & 15;
            int ln  = (i >> 4) & 31;
            int mt  = (i >> 9) & 1;
            int ktl = i >> 10;
            int m   = mt * 16 + (ln & 15);
            int k   = (ck * 8 + ktl) * 32 + a_klocal(e, ln >> 4);
            int rr  = mb * 32 + m;
            apk[i] = (bf16)enc[(size_t)rr * DD + k];
        }
        __syncthreads();
        const v16bf* av = (const v16bf*)apk;
        for (int ktl = 0; ktl < 8; ++ktl) {
            int kt = ck * 8 + ktl;
            v16bf b  = Wvp[((size_t)ntl * 32 + kt) * 32 + lane];
            v16bf a0 = av[(ktl * 2 + 0) * 32 + lane];
            v16bf a1 = av[(ktl * 2 + 1) * 32 + lane];
            c0 = wmma_bf16(a0, b, c0);
            c1 = wmma_bf16(a1, b, c1);
        }
        __syncthreads();
    }
    int n = ntl * 16 + (lane & 15);
    float bbias = (nt < 64 ? bk : bv)[n];
    bf16* outp = (nt < 64) ? Ko : Vo;
    int h = n >> 6, dh = n & 63;
    int mh = (lane >> 4) * 8;
    for (int r = 0; r < 8; ++r) {
        int rr0 = mb * 32 + mh + r;
        int rr1 = rr0 + 16;
        int b0 = rr0 >> 10, s0 = rr0 & 1023;
        int b1 = rr1 >> 10, s1 = rr1 & 1023;
        outp[(((size_t)b0 * HH + h) * SS + s0) * DHH + dh] = (bf16)(c0[r] + bbias);
        outp[(((size_t)b1 * HH + h) * SS + s1) * DHH + dh] = (bf16)(c1[r] + bbias);
    }
}

// ---------------------------------------------------------------------------
// LSTM gates (torch order i,f,g,o). Combines split-K GEMM partials in fixed
// order, adds both biases, updates c in place, and writes h directly as
// WMMA-A packed bf16 into up to two destination A-buffers.
// ---------------------------------------------------------------------------
__global__ void lstm_gates(const float* __restrict__ gp, int ksplit,
                           const float* __restrict__ bih, const float* __restrict__ bhh,
                           float* __restrict__ c,
                           bf16* __restrict__ hdst1, bf16* __restrict__ hdst2) {
    int i = blockIdx.x * blockDim.x + threadIdx.x;
    if (i >= BB * DD) return;
    int b = i >> 10, d = i & 1023;
    float gi = bih[d]          + bhh[d];
    float gf = bih[DD + d]     + bhh[DD + d];
    float gg = bih[2 * DD + d] + bhh[2 * DD + d];
    float go = bih[3 * DD + d] + bhh[3 * DD + d];
    for (int ks = 0; ks < ksplit; ++ks) {
        const float* gr = gp + (size_t)ks * BB * 4 * DD + (size_t)b * 4 * DD;
        gi += gr[d];
        gf += gr[DD + d];
        gg += gr[2 * DD + d];
        go += gr[3 * DD + d];
    }
    float cp = c[i];
    float cn = sigf(gf) * cp + sigf(gi) * tanhf(gg);
    c[i] = cn;
    float hn = sigf(go) * tanhf(cn);
    store_packed(hdst1, b, d, hn);
    if (hdst2) store_packed(hdst2, b, d, hn);
}

// ---------------------------------------------------------------------------
// Cross-attention core, one block per (b,h). Combines q split-K partials +
// bias on load; writes ctx directly as WMMA-A packed bf16 (out_proj input).
// ---------------------------------------------------------------------------
__global__ void attention(const float* __restrict__ qp, int ksplit,
                          const float* __restrict__ bq,
                          const bf16* __restrict__ Ko, const bf16* __restrict__ Vo,
                          bf16* __restrict__ ctxpk) {
    __shared__ float qs[DHH];
    __shared__ float sc[SS];
    __shared__ float red[256];
    __shared__ float cpart[4 * DHH];
    int bh = blockIdx.x;
    int b = bh >> 4, h = bh & 15;
    int tid = threadIdx.x;
    if (tid < DHH) {
        size_t idx = (size_t)b * DD + h * DHH + tid;
        float v = bq[h * DHH + tid];
        for (int ks = 0; ks < ksplit; ++ks) v += qp[(size_t)ks * BB * DD + idx];
        qs[tid] = v;
    }
    __syncthreads();
    const bf16* Kb = Ko + (size_t)bh * SS * DHH;
    float mx = -1e30f;
    for (int s = tid; s < SS; s += 256) {
        const bf16* kr = Kb + (size_t)s * DHH;
        float acc = 0.f;
        for (int d = 0; d < DHH; ++d) acc += qs[d] * (float)kr[d];
        acc *= 0.125f;  // DH^-0.5
        sc[s] = acc;
        mx = fmaxf(mx, acc);
    }
    red[tid] = mx; __syncthreads();
    for (int off = 128; off > 0; off >>= 1) {
        if (tid < off) red[tid] = fmaxf(red[tid], red[tid + off]);
        __syncthreads();
    }
    mx = red[0]; __syncthreads();
    float sm = 0.f;
    for (int s = tid; s < SS; s += 256) { float e = __expf(sc[s] - mx); sc[s] = e; sm += e; }
    red[tid] = sm; __syncthreads();
    for (int off = 128; off > 0; off >>= 1) {
        if (tid < off) red[tid] += red[tid + off];
        __syncthreads();
    }
    float inv = 1.f / red[0];
    const bf16* Vb = Vo + (size_t)bh * SS * DHH;
    int d = tid & 63, sg = tid >> 6;
    float acc = 0.f;
    for (int s = sg; s < SS; s += 4) acc += sc[s] * (float)Vb[(size_t)s * DHH + d];
    cpart[tid] = acc; __syncthreads();
    if (tid < DHH) {
        float v = (cpart[tid] + cpart[tid + 64] + cpart[tid + 128] + cpart[tid + 192]) * inv;
        store_packed(ctxpk, b, h * DHH + tid, v);
    }
}

// ---------------------------------------------------------------------------
// Combine fc2 split-K partials (+bias) -> logits, then log_softmax + greedy
// argmax (first-max tie-break matching jnp.argmax). Writes out[b][t][:].
// ---------------------------------------------------------------------------
__global__ void lsm_argmax(const float* __restrict__ lp, int ksplit,
                           const float* __restrict__ bias,
                           float* __restrict__ logits, float* __restrict__ out,
                           int* __restrict__ tok, int t) {
    __shared__ float rmx[256];
    __shared__ int   rix[256];
    __shared__ float rsm[256];
    int b = blockIdx.x, tid = threadIdx.x;
    float* lrow = logits + (size_t)b * VV;
    float mx = -3.4e38f; int ix = 0;
    for (int v = tid; v < VV; v += 256) {
        float x = bias[v];
        for (int ks = 0; ks < ksplit; ++ks)
            x += lp[(size_t)ks * BB * VV + (size_t)b * VV + v];
        lrow[v] = x;
        if (x > mx) { mx = x; ix = v; }
    }
    rmx[tid] = mx; rix[tid] = ix; __syncthreads();
    for (int off = 128; off > 0; off >>= 1) {
        if (tid < off) {
            float a = rmx[tid], c = rmx[tid + off];
            if (c > a || (c == a && rix[tid + off] < rix[tid])) {
                rmx[tid] = c; rix[tid] = rix[tid + off];
            }
        }
        __syncthreads();
    }
    mx = rmx[0];
    int amax = rix[0];
    __syncthreads();
    float sm = 0.f;
    for (int v = tid; v < VV; v += 256) sm += __expf(lrow[v] - mx);
    rsm[tid] = sm; __syncthreads();
    for (int off = 128; off > 0; off >>= 1) {
        if (tid < off) rsm[tid] += rsm[tid + off];
        __syncthreads();
    }
    float lse = mx + __logf(rsm[0]);
    float* orow = out + ((size_t)b * TT + t) * VV;
    for (int v = tid; v < VV; v += 256) orow[v] = lrow[v] - lse;
    if (tid == 0) tok[b] = amax;
}

// Zero c-state and the packed-h halves used before first production; load
// initial tokens targets[:,0].
__global__ void init_state(const int* __restrict__ targets, int* __restrict__ tok0,
                           float* __restrict__ cst,
                           bf16* __restrict__ a0h, bf16* __restrict__ a1h) {
    int i = blockIdx.x * blockDim.x + threadIdx.x;
    if (i < 2 * BB * DD) cst[i] = 0.f;
    if (i < 1024 * 32) { a0h[i] = (bf16)0.f; a1h[i] = (bf16)0.f; }
    if (i < BB) tok0[i] = targets[(size_t)i * TT];
}

// ---------------------------------------------------------------------------
extern "C" void kernel_launch(void* const* d_in, const int* in_sizes, int n_in,
                              void* d_out, int out_size, void* d_ws, size_t ws_size,
                              hipStream_t stream) {
    const float* enc        = (const float*)d_in[0];
    const int*   targets    = (const int*)d_in[1];
    const float* emb        = (const float*)d_in[2];
    const float* w_ih       = (const float*)d_in[3];
    const float* w_hh       = (const float*)d_in[4];
    const float* b_ih       = (const float*)d_in[5];
    const float* b_hh       = (const float*)d_in[6];
    const float* in_proj_w  = (const float*)d_in[7];
    const float* in_proj_b  = (const float*)d_in[8];
    const float* out_proj_w = (const float*)d_in[9];
    const float* out_proj_b = (const float*)d_in[10];
    const float* fc1_w      = (const float*)d_in[11];
    const float* fc1_b      = (const float*)d_in[12];
    const float* fc2_w      = (const float*)d_in[13];
    const float* fc2_b      = (const float*)d_in[14];
    float* out = (float*)d_out;

    // ---- workspace carve (256B aligned) ----
    char* p = (char*)d_ws;
    auto take = [&](size_t bytes) -> char* {
        char* r = p;
        p += (bytes + 255) & ~(size_t)255;
        return r;
    };
    bf16* wl0 = (bf16*)take((size_t)256 * 64 * 512 * 2);   // LSTM L0 [4096 x 2048]
    bf16* wl1 = (bf16*)take((size_t)256 * 64 * 512 * 2);   // LSTM L1
    bf16* wqp = (bf16*)take((size_t)64 * 32 * 512 * 2);    // wq
    bf16* wkp = (bf16*)take((size_t)64 * 32 * 512 * 2);    // wk
    bf16* wvp = (bf16*)take((size_t)64 * 32 * 512 * 2);    // wv
    bf16* wop = (bf16*)take((size_t)64 * 32 * 512 * 2);    // out_proj
    bf16* wf1 = (bf16*)take((size_t)64 * 64 * 512 * 2);    // fc1 [1024 x 2048]
    bf16* wf2 = (bf16*)take((size_t)625 * 32 * 512 * 2);   // fc2 [10000 x 1024]
    bf16* Kpk = (bf16*)take((size_t)BB * HH * SS * DHH * 2);
    bf16* Vpk = (bf16*)take((size_t)BB * HH * SS * DHH * 2);
    int*   tok = (int*)take(2 * BB * 4);
    float* cst = (float*)take((size_t)2 * BB * DD * 4);    // c state, both layers
    bf16*  A0  = (bf16*)take((size_t)2048 * 32 * 2);       // [emb(tok) | h0]
    bf16*  A1  = (bf16*)take((size_t)2048 * 32 * 2);       // [h0 | h1]
    bf16*  AQ  = (bf16*)take((size_t)1024 * 32 * 2);       // h1 (q proj input)
    bf16*  AO  = (bf16*)take((size_t)1024 * 32 * 2);       // ctx (out_proj input)
    bf16*  AF  = (bf16*)take((size_t)2048 * 32 * 2);       // [h1 | ctxp] (fc1 in)
    bf16*  A2  = (bf16*)take((size_t)1024 * 32 * 2);       // tanh(fc1) (fc2 in)
    float* gp  = (float*)take((size_t)4 * BB * 4 * DD * 4);   // LSTM gate partials
    float* qp  = (float*)take((size_t)2 * BB * DD * 4);       // q partials
    float* lp  = (float*)take((size_t)2 * BB * VV * 4);       // fc2 partials
    float* logits = (float*)take((size_t)BB * VV * 4);

    // ---- weight packing (bf16, WMMA-B layout; L2-resident afterwards) ----
    {
        long tl = (long)256 * 64 * 512;
        pack_b<<<(int)((tl + 255) / 256), 256, 0, stream>>>(w_ih, 1024, w_hh, 1024, wl0, 64, tl);
        pack_b<<<(int)((tl + 255) / 256), 256, 0, stream>>>(w_ih + (size_t)4096 * 1024, 1024,
                                                            w_hh + (size_t)4096 * 1024, 1024, wl1, 64, tl);
        long ts = (long)64 * 32 * 512;
        pack_b<<<(int)((ts + 255) / 256), 256, 0, stream>>>(in_proj_w, 1024, in_proj_w, 0, wqp, 32, ts);
        pack_b<<<(int)((ts + 255) / 256), 256, 0, stream>>>(in_proj_w + (size_t)1024 * 1024, 1024,
                                                            in_proj_w, 0, wkp, 32, ts);
        pack_b<<<(int)((ts + 255) / 256), 256, 0, stream>>>(in_proj_w + (size_t)2048 * 1024, 1024,
                                                            in_proj_w, 0, wvp, 32, ts);
        pack_b<<<(int)((ts + 255) / 256), 256, 0, stream>>>(out_proj_w, 1024, out_proj_w, 0, wop, 32, ts);
        long tf1 = (long)64 * 64 * 512;
        pack_b<<<(int)((tf1 + 255) / 256), 256, 0, stream>>>(fc1_w, 2048, fc1_w, 0, wf1, 64, tf1);
        long tf2 = (long)625 * 32 * 512;
        pack_b<<<(int)((tf2 + 255) / 256), 256, 0, stream>>>(fc2_w, 1024, fc2_w, 0, wf2, 32, tf2);
    }

    // ---- one-time K/V projection (WMMA, LDS-staged) ----
    kv_gemm<<<dim3(BB * SS / 32, 16), 256, 0, stream>>>(
        enc, wkp, wvp, in_proj_b + 1024, in_proj_b + 2048, Kpk, Vpk);

    // ---- init: c=0, zero packed-h halves consumed in step 0, tok0 ----
    init_state<<<(2 * BB * DD + 255) / 256, 256, 0, stream>>>(
        targets, tok, cst, A0 + 1024 * 32, A1 + 1024 * 32);

    float* c0 = cst;
    float* c1 = cst + BB * DD;

    // ---- sequential greedy decode: 11 kernels / step ----
    for (int t = 0; t < TT; ++t) {
        int* tcur  = tok + (t & 1) * BB;
        int* tnext = tok + ((t + 1) & 1) * BB;

        // 1) embedding gather -> A0 first half (h0 half written by prev step)
        pack_a<<<128, 256, 0, stream>>>(emb, tcur, 1024, A0, 1024 * 32);

        // 2-3) LSTM layer 0 (split-K x4); gates pack h0 -> A1[:1024], A0[1024:]
        gemm32<<<dim3(32, 4), 256, 0, stream>>>(A0, wl0, nullptr, gp, nullptr,
                                                4096, 256, 64, 0, 4);
        lstm_gates<<<128, 256, 0, stream>>>(gp, 4, b_ih, b_hh, c0,
                                            A1, A0 + 1024 * 32);

        // 4-5) LSTM layer 1; gates pack h1 -> AQ, AF[:1024]
        gemm32<<<dim3(32, 4), 256, 0, stream>>>(A1, wl1, nullptr, gp, nullptr,
                                                4096, 256, 64, 0, 4);
        lstm_gates<<<128, 256, 0, stream>>>(gp, 4, b_ih + 4096, b_hh + 4096, c1,
                                            AQ, AF);

        // 6) q projection (split-K x2, partials; bias added in attention)
        gemm32<<<dim3(8, 2), 256, 0, stream>>>(AQ, wqp, nullptr, qp, nullptr,
                                               1024, 64, 32, 0, 2);

        // 7) attention core; ctx packed -> AO
        attention<<<BB * HH, 256, 0, stream>>>(qp, 2, in_proj_b, Kpk, Vpk, AO);

        // 8) out_proj; packed output -> AF second half
        gemm32<<<dim3(8, 1), 256, 0, stream>>>(AO, wop, out_proj_b, nullptr,
                                               AF + 1024 * 32, 1024, 64, 32, 0, 1);

        // 9) fc1 + tanh; packed output -> A2
        gemm32<<<dim3(8, 1), 256, 0, stream>>>(AF, wf1, fc1_b, nullptr, A2,
                                               1024, 64, 64, 1, 1);

        // 10) fc2 (split-K x2, partials; bias added in lsm)
        gemm32<<<dim3(79, 2), 256, 0, stream>>>(A2, wf2, nullptr, lp, nullptr,
                                                VV, 625, 32, 0, 2);

        // 11) combine + log_softmax + greedy feedback
        lsm_argmax<<<BB, 256, 0, stream>>>(lp, 2, fc2_b, logits, out, tnext, t);
    }
}
